// VQEmbeddingEMA_10041633538655
// MI455X (gfx1250) — matrix operand order repaired and checked by reading, not compiled, not run on hardware
//
#include <hip/hip_runtime.h>
#include <hip/hip_bf16.h>

// ---------------------------------------------------------------------------
// VQ-VAE vector quantization forward for MI455X (gfx1250, wave32, WMMA+TDM).
//
// argmin_k(||x||^2 - 2x.e_k + ||e_k||^2) == argmin_k(||e_k||^2 - 2 x.e_k):
// dot products via v_wmma_f32_16x16x32_bf16 (fp32 accumulate); the 65536x1024
// distance matrix is never materialized. Codebook tiles are staged into LDS
// by the Tensor Data Mover (tensor_load_to_lds, TENSORcnt, double-buffered)
// so the 8 waves of a block share each 8KB tile instead of each streaming the
// full 512KB bf16 codebook from L2 (8x L2-traffic reduction, DMA overlapped
// with WMMA compute). All 8 B operands of a tile are loaded into distinct
// registers before the WMMA chain so ds_load latency hides under the matrix
// pipe (staggered s_wait_dscnt instead of wait-0 per WMMA).
// ---------------------------------------------------------------------------

typedef __attribute__((ext_vector_type(16))) __bf16        v16bf;
typedef __attribute__((ext_vector_type(8)))  float         v8f;
typedef __attribute__((ext_vector_type(4)))  unsigned int  u32x4;
typedef __attribute__((ext_vector_type(8)))  int           i32x8;
typedef __attribute__((ext_vector_type(4)))  int           i32x4;

#define B_   16
#define S_   4096
#define D_   256
#define K_   1024
#define N_   (B_ * S_)        // 65536 rows
#define QN_  (N_ * D_)        // 16777216 quantized elements
#define STRIPS_ (N_ / 16)     // 4096 row strips (one wave32 each)
#define WPB_ 8                // waves per block (256 threads)
#define TILES_ (K_ / 16)      // 64 code tiles

// LDS tile: 16 code rows, each 256 bf16 (512B) + 16B pad -> bank-conflict-free
#define ROW_BF16_ 264         // 528 bytes per row
#define BUF_BF16_ (16 * ROW_BF16_)

struct BPair { u32x4 lo, hi; };

// ---------------------------------------------------------------------------
// Prep: fp32 codebook -> bf16 copy (WMMA operand precision) + fp32 row norms.
// ---------------------------------------------------------------------------
__global__ __launch_bounds__(256) void vq_prep(const float* __restrict__ E,
                                               __bf16* __restrict__ Ebf,
                                               float* __restrict__ norms) {
  __shared__ float red[256];
  const int k = blockIdx.x;
  const int t = threadIdx.x;
  const float v = E[(size_t)k * D_ + t];
  Ebf[(size_t)k * D_ + t] = (__bf16)v;
  red[t] = v * v;
  __syncthreads();
  for (int s = 128; s > 0; s >>= 1) {
    if (t < s) red[t] += red[t + s];
    __syncthreads();
  }
  if (t == 0) norms[k] = red[0];
}

// ---------------------------------------------------------------------------
// Main fused kernel.
//   A (16x256 bf16) register-resident, contiguous k order (k = 16*half + s), same as B.
//   B tiles DMA'd into LDS by the TDM (double buffer), read as ds_load_b128.
//   C layout: VGPR v, lanes 0-15 -> (row v, col lane); lanes 16-31 ->
//   (row v+8, col lane-16). Per-lane argmin, xor-reduce in 16-lane halves,
//   then fused gather / STE output / index write / loss partial.
// ---------------------------------------------------------------------------
__global__ __launch_bounds__(256) void vq_argmin_gather(
    const float* __restrict__ X, const float* __restrict__ E,
    const __bf16* __restrict__ Ebf, const float* __restrict__ norms,
    float* __restrict__ outQ, float* __restrict__ outIdx,
    float* __restrict__ partials) {
  __shared__ __bf16 ldsB[2][BUF_BF16_];

  const int wave  = threadIdx.x >> 5;
  const int lane  = threadIdx.x & 31;
  const int strip = blockIdx.x * WPB_ + wave;
  const int row0  = strip * 16;
  const int m     = lane & 15;   // row (A) / column (B,C)
  const int khalf = lane >> 4;   // which K-half this lane holds

  // ---- load A tile (16 rows x 256 K) as bf16, resident for all 64 B tiles
  v16bf a[8];
  const float* xrow = X + (size_t)(row0 + m) * D_;
#pragma unroll
  for (int kk = 0; kk < 8; ++kk) {
    const float* p0 = xrow + kk * 32 + khalf * 16;
    const float* p1 = p0 + 8;
    v8f lo = *(const v8f*)p0;
    v8f hi = *(const v8f*)p1;
    v16bf av;
#pragma unroll
    for (int j = 0; j < 8; ++j) {
      av[j]     = (__bf16)lo[j];
      av[8 + j] = (__bf16)hi[j];
    }
    a[kk] = av;
  }

  // ---- TDM descriptor pieces (uniform). Tile = 1-D run of 1024 x 8B elems
  // (16 contiguous code rows), LDS-padded 4 DWORDs every 128 DWORDs so each
  // 512B code row lands on a 528B LDS stride.
  const unsigned int ldsBase = (unsigned int)(size_t)(&ldsB[0][0]);
  const i32x4 z4 = {};
  const i32x8 z8 = {};

  auto tdm_issue = [&](int tile, int buf) {
    const unsigned long long gaddr =
        (unsigned long long)(size_t)(Ebf + (size_t)(tile * 16) * D_);
    u32x4 g0;
    g0[0] = 1u;                                        // count=1 (valid D#)
    g0[1] = ldsBase + (unsigned int)buf * (BUF_BF16_ * 2u);  // lds_addr
    g0[2] = (unsigned int)gaddr;                       // global_addr[31:0]
    g0[3] = (unsigned int)(gaddr >> 32) | (2u << 30);  // addr[56:32] | type=2
    i32x8 g1;
    g1[0] = (int)((3u << 16)     // data_size = 8B
                | (1u << 20)     // pad_enable
                | (6u << 22)     // pad_interval: every 128 DWORDs (512B)
                | (3u << 25));   // pad_amount: 4 DWORDs (16B)
    g1[1] = (int)(1024u << 16);  // tensor_dim0[15:0]   (1024 x 8B)
    g1[2] = (int)(1u << 16);     // tensor_dim1 = 1
    g1[3] = (int)(1024u << 16);  // tile_dim0 = 1024
    g1[4] = 0;                   // tile_dim1/2 unused (1-D tile)
    g1[5] = 1024;                // tensor_dim0_stride[31:0]
    g1[6] = 0;
    g1[7] = 0;
    // 6-arg toolchain form: (g0, g1, grp2, grp3, grp_extra, cpol)
    __builtin_amdgcn_tensor_load_to_lds(g0, g1, z4, z4, z8, 0);
  };

  if (wave == 0) tdm_issue(0, 0);

  float rmin[8];
  int   ridx[8];
#pragma unroll
  for (int v = 0; v < 8; ++v) { rmin[v] = 3.4e38f; ridx[v] = 0; }

  // ---- sweep all 1024 codes, 16 per LDS tile, DMA overlapped with compute
  for (int t = 0; t < TILES_; ++t) {
    if (wave == 0) __builtin_amdgcn_s_wait_tensorcnt(0);  // tile t resident
    __syncthreads();                    // tile t visible; buf (t+1)&1 is free
    if (wave == 0 && (t + 1) < TILES_) tdm_issue(t + 1, (t + 1) & 1);

    const int k0 = t * 16;
    const float nb = norms[k0 + m];     // ||e||^2 bias for my column
    const __bf16* brow = &ldsB[t & 1][0] + m * ROW_BF16_ + khalf * 16;

    // All 8 B operands into distinct registers first: one ds_load clause,
    // staggered dscnt waits, LDS latency overlapped with the WMMA chain.
    v16bf b[8];
#pragma unroll
    for (int kk = 0; kk < 8; ++kk) {
      const u32x4* bp = (const u32x4*)(brow + kk * 32);  // 2 x ds_load_b128
      BPair p{bp[0], bp[1]};
      b[kk] = __builtin_bit_cast(v16bf, p);
    }
    v8f c = {};
#pragma unroll
    for (int kk = 0; kk < 8; ++kk) {
      c = __builtin_amdgcn_wmma_f32_16x16x32_bf16(
              /*neg_a=*/false, a[kk], /*neg_b=*/false, b[kk],
              /*c_mod=*/(short)0, c, /*reuse_a=*/false, /*reuse_b=*/false);
    }
#pragma unroll
    for (int v = 0; v < 8; ++v) {
      const float d = nb - 2.0f * c[v];                  // ||e||^2 - 2 x.e
      if (d < rmin[v]) { rmin[v] = d; ridx[v] = k0 + m; }
    }
  }

  // ---- argmin across the 16 columns held by each lane half (tie -> low idx)
#pragma unroll
  for (int mask = 8; mask >= 1; mask >>= 1) {
#pragma unroll
    for (int v = 0; v < 8; ++v) {
      const float ov = __shfl_xor(rmin[v], mask, 32);
      const int   oi = __shfl_xor(ridx[v], mask, 32);
      if (ov < rmin[v] || (ov == rmin[v] && oi < ridx[v])) {
        rmin[v] = ov; ridx[v] = oi;
      }
    }
  }

  // ---- epilogue: gather fp32 code rows (STE forward == gathered embedding),
  //      write indices, accumulate commitment-loss partial.
  float lsum = 0.0f;
#pragma unroll
  for (int r = 0; r < 16; ++r) {
    const int src = (r < 8) ? 0 : 16;
    const int vv  = (r < 8) ? r : (r - 8);
    const int kr  = __shfl(ridx[vv], src, 32);
    const float* er = E + (size_t)kr * D_;
    const float* xr = X + (size_t)(row0 + r) * D_;
    float* qr = outQ + (size_t)(row0 + r) * D_;
#pragma unroll
    for (int c8 = 0; c8 < 8; ++c8) {
      const int ci = c8 * 32 + lane;          // coalesced 128B per wave
      const float q  = er[ci];
      const float xv = xr[ci];
      qr[ci] = q;
      const float df = q - xv;
      lsum += df * df;
    }
    if (lane == 0) outIdx[row0 + r] = (float)kr;
  }
#pragma unroll
  for (int mask = 16; mask >= 1; mask >>= 1) lsum += __shfl_xor(lsum, mask, 32);
  if (lane == 0) partials[strip] = lsum;
}

// ---------------------------------------------------------------------------
// Deterministic loss reduction (no fp atomics -> identical on every replay).
// ---------------------------------------------------------------------------
__global__ __launch_bounds__(256) void vq_loss(const float* __restrict__ partials,
                                               float* __restrict__ lossOut) {
  __shared__ float red[256];
  const int t = threadIdx.x;
  float s = 0.0f;
  for (int i = t; i < STRIPS_; i += 256) s += partials[i];
  red[t] = s;
  __syncthreads();
  for (int k = 128; k > 0; k >>= 1) {
    if (t < k) red[t] += red[t + k];
    __syncthreads();
  }
  if (t == 0) lossOut[0] = 0.25f * red[0] / (float)QN_;
}

extern "C" void kernel_launch(void* const* d_in, const int* in_sizes, int n_in,
                              void* d_out, int out_size, void* d_ws, size_t ws_size,
                              hipStream_t stream) {
  const float* X = (const float*)d_in[0];   // [B,S,D] fp32
  const float* E = (const float*)d_in[1];   // [K,D]  fp32

  float* out     = (float*)d_out;
  float* outQ    = out;                     // [B,S,D]
  float* lossOut = out + QN_;               // scalar
  float* outIdx  = out + QN_ + 1;           // [B,S] (as float values)

  char*   ws       = (char*)d_ws;
  __bf16* Ebf      = (__bf16*)ws;                              // 512 KB
  float*  norms    = (float*)(ws + (size_t)K_ * D_ * 2);       // 4 KB
  float*  partials = norms + K_;                               // 16 KB

  vq_prep<<<K_, 256, 0, stream>>>(E, Ebf, norms);
  vq_argmin_gather<<<STRIPS_ / WPB_, 256, 0, stream>>>(
      X, E, Ebf, norms, outQ, outIdx, partials);
  vq_loss<<<1, 256, 0, stream>>>(partials, lossOut);
}
